// AttnBlock_68212670595740
// MI455X (gfx1250) — matrix operand order repaired and checked by reading, not compiled, not run on hardware
//
#include <hip/hip_runtime.h>
#include <hip/hip_bf16.h>

// ---------------------------------------------------------------------------
// MI455X (gfx1250) fused attention block.
//  - All matmuls via v_wmma_f32_16x16x32_bf16 (fp32 accumulate).
//  - Flash-attention fusion: N x N scores never hit HBM (saves ~268 MB/round).
//  - K/V tiles double-buffered in LDS via GLOBAL_LOAD_ASYNC_TO_LDS_B128
//    (ASYNCcnt-tracked), overlapping global latency with WMMA compute.
//  - Q tile register-resident (16 A-fragments/wave = 128 VGPRs).
// ---------------------------------------------------------------------------

typedef __attribute__((ext_vector_type(16))) __bf16 v16bf;
typedef __attribute__((ext_vector_type(8)))  __bf16 v8bf;
typedef __attribute__((ext_vector_type(8)))  float  v8f;

#define WMMA_BF16(a, b, c) \
  __builtin_amdgcn_wmma_f32_16x16x32_bf16(false, (a), false, (b), (short)0, (c), false, false)

static constexpr int C   = 512;
static constexpr int N   = 4096;       // 64*64
static constexpr int GRP = 32;
static constexpr int CG  = 16;         // channels per group

// --- WMMA fragment loaders --------------------------------------------------
// A (16x32 bf16, MxK): lane<16 holds row=lane, K = [k0..k0+15]
// with k0 = kk (+16 for lanes>=16). Rows contiguous in memory (row stride ld).
__device__ inline v16bf load_a_bf16(const __bf16* base, int ld, int kk) {
  const int lane = threadIdx.x & 31;
  const int row  = lane & 15;
  const int k0   = kk + ((lane >> 4) << 4);
  const __bf16* p = base + (size_t)row * ld + k0;
  union { v16bf v; v8bf h[2]; } u;
  u.h[0] = *(const v8bf*)(p);
  u.h[1] = *(const v8bf*)(p + 8);
  return u.v;
}

// B (32x16 bf16, KxN): lane<16 holds col=lane, K=[kk..kk+15]; lanes>=16 K+=16.
// Requires element B[k][col] contiguous in k, i.e. source stored [col][k].
__device__ inline v16bf load_b_bf16(const __bf16* base, int ld, int kk) {
  const int lane = threadIdx.x & 31;
  const int col  = lane & 15;
  const int k0   = kk + ((lane >> 4) << 4);
  return *(const v16bf*)(base + (size_t)col * ld + k0);
}

// Async global->LDS copy of 16 bytes per lane (CDNA5 async path, ASYNCcnt).
__device__ inline void async_ld16(unsigned lds_addr, const void* gaddr) {
  asm volatile("global_load_async_to_lds_b128 %0, %1, off"
               :: "v"(lds_addr), "v"(gaddr) : "memory");
}

// --- Kernel 1: GroupNorm, fp32 in -> bf16 hT [B, N, C] (transposed) ---------
__global__ __launch_bounds__(256)
void gn_kernel(const float* __restrict__ x, const float* __restrict__ gamma,
               const float* __restrict__ beta, __bf16* __restrict__ hT) {
  const int bg  = blockIdx.x;
  const int b   = bg >> 5, grp = bg & 31;
  const int t   = threadIdx.x;
  const float* xg = x + ((size_t)b * C + (size_t)grp * CG) * N;

  float s = 0.f, ss = 0.f;
  for (int c = 0; c < CG; ++c) {
    const float4* row = (const float4*)(xg + (size_t)c * N);
    for (int i = t; i < N / 4; i += 256) {
      float4 v = row[i];
      s  += v.x + v.y + v.z + v.w;
      ss += v.x * v.x + v.y * v.y + v.z * v.z + v.w * v.w;
    }
  }
  __shared__ float rs[256], rss[256];
  rs[t] = s; rss[t] = ss;
  __syncthreads();
  for (int off = 128; off > 0; off >>= 1) {
    if (t < off) { rs[t] += rs[t + off]; rss[t] += rss[t + off]; }
    __syncthreads();
  }
  const float inv_n = 1.f / (float)(CG * N);
  const float mean  = rs[0] * inv_n;
  const float var   = rss[0] * inv_n - mean * mean;
  const float rstd  = rsqrtf(var + 1e-6f);

  float ga[CG], bb[CG];
#pragma unroll
  for (int c = 0; c < CG; ++c) {
    ga[c] = gamma[grp * CG + c] * rstd;
    bb[c] = beta[grp * CG + c] - mean * ga[c];
  }
  // Transposed write: lane-consecutive n -> 32B-contiguous bf16 row chunks.
  for (int n = t; n < N; n += 256) {
    __bf16 o[CG];
#pragma unroll
    for (int c = 0; c < CG; ++c)
      o[c] = (__bf16)(xg[(size_t)c * N + n] * ga[c] + bb[c]);
    __bf16* dst = hT + ((size_t)b * N + n) * C + grp * CG;
    *(v16bf*)dst = *(v16bf*)o;
  }
}

// --- Kernel 2: fp32 -> bf16 weight pre-convert ------------------------------
__global__ void wcvt_kernel(const float* __restrict__ w, __bf16* __restrict__ o,
                            int nelem) {
  for (int i = blockIdx.x * blockDim.x + threadIdx.x; i < nelem;
       i += gridDim.x * blockDim.x)
    o[i] = (__bf16)w[i];
}

// --- Kernel 3: fused QKV. hT [N,C] -> qT [N,C] (scaled), kT [N,C], v [C,N] --
// Block tile: 128 n x 128 o; 8 waves: (wn=wid&3: 32-row n band, wo=wid>>2:
// 64-col o band). q/k as D[n,o]; v as D[o,n] so all stores are coalesced and
// v lands in the [C,N] layout flash attention needs for its B fragments.
__global__ __launch_bounds__(256)
void qkv_kernel(const __bf16* __restrict__ hT,
                const __bf16* __restrict__ wq, const float* __restrict__ bq,
                const __bf16* __restrict__ wk, const float* __restrict__ bk,
                const __bf16* __restrict__ wv, const float* __restrict__ bv,
                __bf16* __restrict__ qT, __bf16* __restrict__ kT,
                __bf16* __restrict__ vN) {
  const int b    = blockIdx.z;
  const int nb   = blockIdx.x * 128;
  const int ob   = blockIdx.y * 128;
  const int wid  = threadIdx.x >> 5, lane = threadIdx.x & 31;
  const int wn   = wid & 3, wo = wid >> 2;
  const __bf16* hTb = hT + (size_t)b * N * C;

  v8f qa[2][4] = {}, ka[2][4] = {}, va[4][2] = {};

  for (int kk = 0; kk < C; kk += 32) {
    v16bf a[2], bh[2];
#pragma unroll
    for (int r = 0; r < 2; ++r)
      a[r] = load_a_bf16(hTb + (size_t)(nb + wn * 32 + r * 16) * C, C, kk);
#pragma unroll
    for (int cn = 0; cn < 2; ++cn)   // hT rows as B fragments (for v)
      bh[cn] = load_b_bf16(hTb + (size_t)(nb + wn * 32 + cn * 16) * C, C, kk);
#pragma unroll
    for (int ct = 0; ct < 4; ++ct) { // weight rows o contiguous in c
      v16bf bwq = load_b_bf16(wq + (size_t)(ob + wo * 64 + ct * 16) * C, C, kk);
      v16bf bwk = load_b_bf16(wk + (size_t)(ob + wo * 64 + ct * 16) * C, C, kk);
#pragma unroll
      for (int r = 0; r < 2; ++r) {
        qa[r][ct] = WMMA_BF16(a[r], bwq, qa[r][ct]);
        ka[r][ct] = WMMA_BF16(a[r], bwk, ka[r][ct]);
      }
    }
#pragma unroll
    for (int rt = 0; rt < 4; ++rt) {
      v16bf awv = load_a_bf16(wv + (size_t)(ob + wo * 64 + rt * 16) * C, C, kk);
#pragma unroll
      for (int cn = 0; cn < 2; ++cn)
        va[rt][cn] = WMMA_BF16(awv, bh[cn], va[rt][cn]);
    }
  }

  const int   col = lane & 15, rb = (lane >> 4) << 3;
  const float qscale = 0.04419417382415922f;  // 512^-0.5 folded into q
#pragma unroll
  for (int r = 0; r < 2; ++r)
#pragma unroll
    for (int ct = 0; ct < 4; ++ct) {
      const int   o  = ob + wo * 64 + ct * 16 + col;
      const float biq = bq[o], bik = bk[o];
#pragma unroll
      for (int e = 0; e < 8; ++e) {
        const int    n   = nb + wn * 32 + r * 16 + rb + e;
        const size_t idx = ((size_t)b * N + n) * C + o;
        qT[idx] = (__bf16)((qa[r][ct][e] + biq) * qscale);
        kT[idx] = (__bf16)(ka[r][ct][e] + bik);
      }
    }
#pragma unroll
  for (int rt = 0; rt < 4; ++rt)
#pragma unroll
    for (int cn = 0; cn < 2; ++cn) {
      const int n = nb + wn * 32 + cn * 16 + col;
#pragma unroll
      for (int e = 0; e < 8; ++e) {
        const int o = ob + wo * 64 + rt * 16 + rb + e;
        vN[((size_t)b * C + o) * N + n] = (__bf16)(va[rt][cn][e] + bv[o]);
      }
    }
}

// --- Kernel 4: flash attention, async double-buffered K/V -------------------
// 32 queries/block; 8 waves: wqg = wid&1 (16-query group), wc = wid>>1
// (128-channel slice of O). Q is register-resident (16 A-frags). K/V tiles
// (32 keys wide) stream through 2x LDS buffers via async loads.
// LDS: Ks 2x[32][512] (64KB) + Vs 2x[512][32] (64KB) + 8x[16][32] P (8KB).
__global__ __launch_bounds__(256)
void attn_kernel(const __bf16* __restrict__ qT, const __bf16* __restrict__ kT,
                 const __bf16* __restrict__ vN, __bf16* __restrict__ oT) {
  const int b   = blockIdx.y;
  const int qb  = blockIdx.x * 32;
  const int tid = threadIdx.x, lane = tid & 31, wid = tid >> 5;
  const int wqg = wid & 1, wc = wid >> 1;

  extern __shared__ char smem[];
  __bf16* KsBase = (__bf16*)smem;              // 2 x 32 x 512
  __bf16* VsBase = KsBase + 2 * 32 * C;        // 2 x 512 x 32
  __bf16* Ps     = VsBase + 2 * C * 32 + wid * 512;  // per-wave [16][32]
  // LDS byte offsets for async destinations (low 32 bits of the flat shared pointer).
  const unsigned ksOff = (unsigned)(uintptr_t)KsBase;
  const unsigned vsOff = (unsigned)(uintptr_t)VsBase;

  const __bf16* qTb = qT + (size_t)b * N * C;
  const __bf16* kTb = kT + (size_t)b * N * C;
  const __bf16* vNb = vN + (size_t)b * C * N;

  // Register-resident Q: 16 A-fragments covering K=512 for this wave's rows.
  v16bf qf[16];
#pragma unroll
  for (int ks = 0; ks < 16; ++ks)
    qf[ks] = load_a_bf16(qTb + (size_t)(qb + wqg * 16) * C, C, ks * 32);

  v8f   oacc[8] = {};
  float mrun[8], lrun[8];
#pragma unroll
  for (int e = 0; e < 8; ++e) { mrun[e] = -1e30f; lrun[e] = 0.f; }

  // Issue one K/V tile: 16 async b128 ops per thread (8 K + 8 V).
  auto issue_tile = [&](int t) {
    const int      jb  = t * 32;
    const unsigned buf = (unsigned)(t & 1) * 32768u;  // 32KB per buffer
    const char* ksrc = (const char*)(kTb + (size_t)jb * C);  // 32 rows, flat
#pragma unroll
    for (int i = 0; i < 8; ++i) {
      const int idx = tid + i * 256;                  // 16B granule
      async_ld16(ksOff + buf + idx * 16, ksrc + (size_t)idx * 16);
    }
#pragma unroll
    for (int i = 0; i < 8; ++i) {                     // V: 64B per channel row
      const int g = tid + i * 256, c = g >> 2, part = g & 3;
      const char* vsrc = (const char*)(vNb + (size_t)c * N + jb) + part * 16;
      async_ld16(vsOff + buf + g * 16, vsrc);
    }
  };

  issue_tile(0);

  for (int t = 0; t < N / 32; ++t) {
    if (t + 1 < N / 32) {
      issue_tile(t + 1);  // prefetch next tile into the other buffer
      // 16 newest (next-tile) ops may remain in flight; in-order retirement
      // guarantees the current tile's 16 ops are complete.
      asm volatile("s_wait_asynccnt 0x10" ::: "memory");
    } else {
      asm volatile("s_wait_asynccnt 0x0" ::: "memory");
    }
    __syncthreads();

    const __bf16* Ks = KsBase + (size_t)(t & 1) * (32 * C);
    const __bf16* Vs = VsBase + (size_t)(t & 1) * (C * 32);

    // S[16x32] = Q_tile . K_tile^T  (scale pre-folded into qT)
    v8f s0 = {}, s1 = {};
#pragma unroll
    for (int ks = 0; ks < 16; ++ks) {
      v16bf b0 = load_b_bf16(Ks, C, ks * 32);
      v16bf b1 = load_b_bf16(Ks + 16 * C, C, ks * 32);
      s0 = WMMA_BF16(qf[ks], b0, s0);
      s1 = WMMA_BF16(qf[ks], b1, s1);
    }

    // Online softmax: rows live as (vgpr e, lane half); cols across 16 lanes,
    // so xor-shuffles (<16) stay within each half-wave.
    float p0[8], p1[8], alpha[8];
#pragma unroll
    for (int e = 0; e < 8; ++e) {
      float mx = fmaxf(s0[e], s1[e]);
#pragma unroll
      for (int m = 1; m < 16; m <<= 1) mx = fmaxf(mx, __shfl_xor(mx, m, 32));
      const float mn = fmaxf(mrun[e], mx);
      alpha[e] = __expf(mrun[e] - mn);
      p0[e]    = __expf(s0[e] - mn);
      p1[e]    = __expf(s1[e] - mn);
      float rsum = p0[e] + p1[e];
#pragma unroll
      for (int m = 1; m < 16; m <<= 1) rsum += __shfl_xor(rsum, m, 32);
      lrun[e] = lrun[e] * alpha[e] + rsum;
      mrun[e] = mn;
    }
#pragma unroll
    for (int u = 0; u < 8; ++u)
#pragma unroll
      for (int e = 0; e < 8; ++e) oacc[u][e] *= alpha[e];

    {  // P: D layout -> LDS (row major 16x32) -> reread as A fragment
      const int colw = lane & 15, rb = (lane >> 4) << 3;
#pragma unroll
      for (int e = 0; e < 8; ++e) {
        Ps[(rb + e) * 32 + colw]      = (__bf16)p0[e];
        Ps[(rb + e) * 32 + 16 + colw] = (__bf16)p1[e];
      }
    }
    asm volatile("s_wait_dscnt 0x0" ::: "memory");  // wave-local LDS RAW
    const v16bf pa = load_a_bf16(Ps, 32, 0);
#pragma unroll
    for (int u = 0; u < 8; ++u) {
      v16bf bv = load_b_bf16(Vs + (size_t)(wc * 128 + u * 16) * 32, 32, 0);
      oacc[u] = WMMA_BF16(pa, bv, oacc[u]);
    }
    __syncthreads();  // all waves done with this buffer before it is refilled
  }

  const int colw = lane & 15, rb = (lane >> 4) << 3;
#pragma unroll
  for (int e = 0; e < 8; ++e) {
    const float inv = 1.f / lrun[e];
    const int   n   = qb + wqg * 16 + rb + e;
#pragma unroll
    for (int u = 0; u < 8; ++u) {
      const int c = wc * 128 + u * 16 + colw;
      oT[((size_t)b * N + n) * C + c] = (__bf16)(oacc[u][e] * inv);
    }
  }
}

// --- Kernel 5: projection + bias + residual, fp32 out [B, C, N] -------------
// Block tile 128 o x 64 n; waves: wo=wid&3 (32-row o band), wn=wid>>2.
__global__ __launch_bounds__(256)
void proj_kernel(const __bf16* __restrict__ oT, const __bf16* __restrict__ wp,
                 const float* __restrict__ bp, const float* __restrict__ x,
                 float* __restrict__ out) {
  const int b   = blockIdx.z;
  const int nb  = blockIdx.x * 64;
  const int ob  = blockIdx.y * 128;
  const int wid = threadIdx.x >> 5, lane = threadIdx.x & 31;
  const int wo  = wid & 3, wn = wid >> 2;
  const __bf16* oTb = oT + (size_t)b * N * C;

  v8f acc[2][2] = {};
  for (int kk = 0; kk < C; kk += 32) {
    v16bf a[2], bb[2];
#pragma unroll
    for (int r = 0; r < 2; ++r)
      a[r] = load_a_bf16(wp + (size_t)(ob + wo * 32 + r * 16) * C, C, kk);
#pragma unroll
    for (int cn = 0; cn < 2; ++cn)
      bb[cn] = load_b_bf16(oTb + (size_t)(nb + wn * 32 + cn * 16) * C, C, kk);
#pragma unroll
    for (int r = 0; r < 2; ++r)
#pragma unroll
      for (int cn = 0; cn < 2; ++cn)
        acc[r][cn] = WMMA_BF16(a[r], bb[cn], acc[r][cn]);
  }

  const int colw = lane & 15, rb = (lane >> 4) << 3;
#pragma unroll
  for (int r = 0; r < 2; ++r)
#pragma unroll
    for (int cn = 0; cn < 2; ++cn)
#pragma unroll
      for (int e = 0; e < 8; ++e) {
        const int    o   = ob + wo * 32 + r * 16 + rb + e;
        const int    n   = nb + wn * 32 + cn * 16 + colw;
        const size_t idx = ((size_t)b * C + o) * N + n;
        out[idx] = acc[r][cn][e] + bp[o] + x[idx];
      }
}

// ---------------------------------------------------------------------------
extern "C" void kernel_launch(void* const* d_in, const int* in_sizes, int n_in,
                              void* d_out, int out_size, void* d_ws, size_t ws_size,
                              hipStream_t stream) {
  const float* x     = (const float*)d_in[0];
  const float* gamma = (const float*)d_in[1];
  const float* beta  = (const float*)d_in[2];
  const float* wq    = (const float*)d_in[3];
  const float* bq    = (const float*)d_in[4];
  const float* wk    = (const float*)d_in[5];
  const float* bk    = (const float*)d_in[6];
  const float* wv    = (const float*)d_in[7];
  const float* bv    = (const float*)d_in[8];
  const float* wp    = (const float*)d_in[9];
  const float* bp    = (const float*)d_in[10];
  float* out = (float*)d_out;

  const int B = in_sizes[0] / (C * N);   // 4 for the reference shapes
  const size_t tens = (size_t)B * N * C; // elements per [B,N,C] tensor

  // Workspace layout (bf16): hT | qT | kT | v | wqb | wkb | wvb | wpb (~66MB)
  __bf16* hT  = (__bf16*)d_ws;
  __bf16* qT  = hT + tens;
  __bf16* kT  = qT + tens;
  __bf16* vN  = kT + tens;
  __bf16* wqb = vN + tens;
  __bf16* wkb = wqb + (size_t)C * C;
  __bf16* wvb = wkb + (size_t)C * C;
  __bf16* wpb = wvb + (size_t)C * C;
  __bf16* oT  = hT;  // hT is dead after qkv_kernel; reuse for attn output

  gn_kernel<<<dim3(B * GRP), 256, 0, stream>>>(x, gamma, beta, hT);

  wcvt_kernel<<<dim3(64), 256, 0, stream>>>(wq, wqb, C * C);
  wcvt_kernel<<<dim3(64), 256, 0, stream>>>(wk, wkb, C * C);
  wcvt_kernel<<<dim3(64), 256, 0, stream>>>(wv, wvb, C * C);
  wcvt_kernel<<<dim3(64), 256, 0, stream>>>(wp, wpb, C * C);

  qkv_kernel<<<dim3(N / 128, C / 128, B), 256, 0, stream>>>(
      hT, wqb, bq, wkb, bk, wvb, bv, qT, kT, vN);

  // LDS: 2x32KB K + 2x32KB V + 8KB P staging = 136KB (2 blocks / 320KB WGP)
  const size_t smem = (size_t)(2 * 32 * C + 2 * C * 32 + 8 * 16 * 32) * sizeof(__bf16);
  attn_kernel<<<dim3(N / 32, B), 256, smem, stream>>>(qT, kT, vN, oT);

  proj_kernel<<<dim3(N / 64, C / 128, B), 256, 0, stream>>>(oT, wpb, bp, x, out);
}
